// NHP_34454227648647
// MI455X (gfx1250) — matrix operand order repaired and checked by reading, not compiled, not run on hardware
//
#include <hip/hip_runtime.h>
#include <hip/hip_bf16.h>
#include <math.h>

typedef __attribute__((ext_vector_type(16))) _Float16 v16h;
typedef __attribute__((ext_vector_type(8)))  _Float16 v8h;
typedef __attribute__((ext_vector_type(2)))  _Float16 v2h;
typedef __attribute__((ext_vector_type(8)))  float    v8f;
typedef __attribute__((ext_vector_type(4)))  float    v4f;

#define N_NODES 20000
#define N_EDGES 8000
#define KMEM    32
#define FEATD   256
#define EMB     128
#define CONVD   128
#define NBLOCK  (N_NODES / KMEM)   // 625

// ---------------------------------------------------------------------------
// Stage 0: stream the 640MB incidence matrix once (NT loads), scatter member
// ids. inc is row-major [N, E]; member slot j of edge e sits in node block j.
// ---------------------------------------------------------------------------
__global__ __launch_bounds__(256) void nhp_extract_members(
    const float* __restrict__ inc, int* __restrict__ members) {
  const long long total4 = (long long)N_NODES * N_EDGES / 4;
  const long long stride = (long long)gridDim.x * blockDim.x;
  for (long long i = (long long)blockIdx.x * blockDim.x + threadIdx.x;
       i < total4; i += stride) {
    v4f v = __builtin_nontemporal_load((const v4f*)inc + i);
    long long base = i * 4;                 // E % 4 == 0 -> one row per vec4
    int n = (int)(base / N_EDGES);
    int e = (int)(base % N_EDGES);
    int j = n / NBLOCK;
#pragma unroll
    for (int c = 0; c < 4; ++c) {
      if (v[c] != 0.0f) members[(e + c) * KMEM + j] = n;
    }
  }
}

// ---------------------------------------------------------------------------
// Stage 0b: f16 copies of W_rel and (W_root - W_rel)
// ---------------------------------------------------------------------------
__global__ __launch_bounds__(256) void nhp_prep_weights(
    const float* __restrict__ W_rel, const float* __restrict__ W_root,
    _Float16* __restrict__ Wd, _Float16* __restrict__ Wrel) {
  int i = blockIdx.x * blockDim.x + threadIdx.x;
  if (i < CONVD * EMB) {
    Wd[i]   = (_Float16)(W_root[i] - W_rel[i]);
    Wrel[i] = (_Float16)W_rel[i];
  }
}

// ---------------------------------------------------------------------------
// Stage 1: x = f16(feature @ W_enc^T + b_enc)   [N,EMB], WMMA f16, f32 accum.
// One block = 16 rows; 8 waves, wave w owns output columns 16w..16w+15.
// ---------------------------------------------------------------------------
__global__ __launch_bounds__(256) void nhp_encoder(
    const float* __restrict__ feature, const float* __restrict__ W_enc,
    const float* __restrict__ b_enc, _Float16* __restrict__ x) {
  __shared__ _Float16 Atile[16 * 264];     // 16 rows x 256 halfs, +8 pad
  const int tid = threadIdx.x;
  const int mbase = blockIdx.x * 16;

  {  // stage + convert 16x256 f32 -> f16 LDS (each thread 16 floats)
    int row = tid >> 4;
    int col = (tid & 15) * 16;
    const float* src = feature + (long long)(mbase + row) * FEATD + col;
    _Float16* dst = &Atile[row * 264 + col];
#pragma unroll
    for (int q = 0; q < 4; ++q) {
      v4f f = *(const v4f*)(src + q * 4);
#pragma unroll
      for (int c = 0; c < 4; ++c) dst[q * 4 + c] = (_Float16)f[c];
    }
  }
  __syncthreads();

  const int lane = tid & 31, wv = tid >> 5;
  const int lh = lane & 15, hf = lane >> 4;
  const int ncol = wv * 16 + lh;
  v8f acc = {};
#pragma unroll
  for (int k0 = 0; k0 < FEATD; k0 += 32) {
    v8h alo = *(const v8h*)&Atile[lh * 264 + k0 + hf * 16];
    v8h ahi = *(const v8h*)&Atile[lh * 264 + k0 + hf * 16 + 8];
    v16h a = __builtin_shufflevector(alo, ahi, 0,1,2,3,4,5,6,7,8,9,10,11,12,13,14,15);
    const float* wrow = W_enc + (long long)ncol * FEATD + k0 + hf * 16;
    v16h b;
#pragma unroll
    for (int c = 0; c < 16; ++c) b[c] = (_Float16)wrow[c];
    acc = __builtin_amdgcn_wmma_f32_16x16x32_f16(false, a, false, b,
                                                 (short)0, acc, false, false);
  }
  const float bias = b_enc[ncol];
#pragma unroll
  for (int g = 0; g < 8; ++g) {
    int row = mbase + hf * 8 + g;
    x[(long long)row * EMB + ncol] = (_Float16)(acc[g] + bias);
  }
}

// ---------------------------------------------------------------------------
// Stage 2: one block per hyperedge.
//   h = relu( xg @ (W_root-W_rel)^T + (sum xg) @ W_rel^T + b_rel )
//   out = sigmoid( (max_k h - min_k h) . W_out + b_out )
// ---------------------------------------------------------------------------
__global__ __launch_bounds__(256) void nhp_edge(
    const _Float16* __restrict__ x, const int* __restrict__ members,
    const _Float16* __restrict__ Wd, const _Float16* __restrict__ Wrel,
    const float* __restrict__ b_rel, const float* __restrict__ W_out,
    const float* __restrict__ b_out, float* __restrict__ out) {
  __shared__ _Float16 xg[32 * 136];        // 32 rows x 128 halfs, +8 pad
  __shared__ float S[EMB];
  __shared__ float tb[CONVD];
  __shared__ float red[8];
  const int e = blockIdx.x;
  const int tid = threadIdx.x;

  {  // gather member rows (f16, L2-resident x) into LDS
    int chunk = tid & 15;
#pragma unroll
    for (int it = 0; it < 2; ++it) {
      int row = (tid >> 4) + it * 16;
      int node = members[e * KMEM + row];
      v8h v = *(const v8h*)(x + (long long)node * EMB + chunk * 8);
      *(v8h*)&xg[row * 136 + chunk * 8] = v;
    }
  }
  __syncthreads();

  if (tid < 64) {  // column sums S[n] over 32 members (pair per thread)
    float s0 = 0.f, s1 = 0.f;
#pragma unroll
    for (int r = 0; r < 32; ++r) {
      v2h p = *(const v2h*)&xg[r * 136 + 2 * tid];
      s0 += (float)p[0];
      s1 += (float)p[1];
    }
    S[2 * tid] = s0;
    S[2 * tid + 1] = s1;
  }
  __syncthreads();

  if (tid < CONVD) {  // tb[n] = S . W_rel[n,:] + b_rel[n]
    float t = 0.f;
    const _Float16* wr = Wrel + tid * EMB;
#pragma unroll 4
    for (int k2 = 0; k2 < EMB; k2 += 8) {
      v8h w = *(const v8h*)(wr + k2);
#pragma unroll
      for (int c = 0; c < 8; ++c) t += S[k2 + c] * (float)w[c];
    }
    tb[tid] = t + b_rel[tid];
  }
  __syncthreads();

  const int lane = tid & 31, wv = tid >> 5;
  const int lh = lane & 15, hf = lane >> 4;
  const int ncol = wv * 16 + lh;
  v8f c0 = {}, c1 = {};
#pragma unroll
  for (int k0 = 0; k0 < EMB; k0 += 32) {
    v16h b = *(const v16h*)(Wd + (long long)ncol * EMB + k0 + hf * 16);
    v8h a0lo = *(const v8h*)&xg[lh * 136 + k0 + hf * 16];
    v8h a0hi = *(const v8h*)&xg[lh * 136 + k0 + hf * 16 + 8];
    v8h a1lo = *(const v8h*)&xg[(16 + lh) * 136 + k0 + hf * 16];
    v8h a1hi = *(const v8h*)&xg[(16 + lh) * 136 + k0 + hf * 16 + 8];
    v16h a0 = __builtin_shufflevector(a0lo, a0hi, 0,1,2,3,4,5,6,7,8,9,10,11,12,13,14,15);
    v16h a1 = __builtin_shufflevector(a1lo, a1hi, 0,1,2,3,4,5,6,7,8,9,10,11,12,13,14,15);
    c0 = __builtin_amdgcn_wmma_f32_16x16x32_f16(false, a0, false, b,
                                                (short)0, c0, false, false);
    c1 = __builtin_amdgcn_wmma_f32_16x16x32_f16(false, a1, false, b,
                                                (short)0, c1, false, false);
  }

  // epilogue: relu, per-column max/min over the 32 rows, diff . W_out
  const float tbn = tb[ncol];
  float mx = -1e30f, mn = 1e30f;
#pragma unroll
  for (int g = 0; g < 8; ++g) {
    float h0 = c0[g] + tbn; h0 = h0 > 0.f ? h0 : 0.f;
    float h1 = c1[g] + tbn; h1 = h1 > 0.f ? h1 : 0.f;
    mx = fmaxf(mx, fmaxf(h0, h1));
    mn = fminf(mn, fminf(h0, h1));
  }
  mx = fmaxf(mx, __shfl_xor(mx, 16, 32));   // other 16 rows live in lane^16
  mn = fminf(mn, __shfl_xor(mn, 16, 32));
  float partial = (mx - mn) * W_out[ncol];
  partial += __shfl_xor(partial, 8, 32);
  partial += __shfl_xor(partial, 4, 32);
  partial += __shfl_xor(partial, 2, 32);
  partial += __shfl_xor(partial, 1, 32);
  if (lane == 0) red[wv] = partial;
  __syncthreads();
  if (tid == 0) {
    float z = b_out[0];
#pragma unroll
    for (int w = 0; w < 8; ++w) z += red[w];
    out[e] = 1.0f / (1.0f + expf(-z));
  }
}

// ---------------------------------------------------------------------------
extern "C" void kernel_launch(void* const* d_in, const int* in_sizes, int n_in,
                              void* d_out, int out_size, void* d_ws, size_t ws_size,
                              hipStream_t stream) {
  const float* feature = (const float*)d_in[0];
  const float* inc     = (const float*)d_in[1];
  const float* W_enc   = (const float*)d_in[2];
  const float* b_enc   = (const float*)d_in[3];
  const float* W_rel   = (const float*)d_in[4];
  const float* b_rel   = (const float*)d_in[5];
  const float* W_root  = (const float*)d_in[6];
  const float* W_out   = (const float*)d_in[7];
  const float* b_out   = (const float*)d_in[8];
  float* out = (float*)d_out;

  char* ws = (char*)d_ws;
  _Float16* x    = (_Float16*)(ws);                          // 5,120,000 B
  int*      mem  = (int*)     (ws + 5120000);                // 1,024,000 B
  _Float16* Wd   = (_Float16*)(ws + 6144000);                //    32,768 B
  _Float16* Wrel = (_Float16*)(ws + 6176768);                //    32,768 B

  nhp_extract_members<<<4096, 256, 0, stream>>>(inc, mem);
  nhp_prep_weights<<<(CONVD * EMB + 255) / 256, 256, 0, stream>>>(W_rel, W_root, Wd, Wrel);
  nhp_encoder<<<N_NODES / 16, 256, 0, stream>>>(feature, W_enc, b_enc, x);
  nhp_edge<<<N_EDGES, 256, 0, stream>>>(x, mem, Wd, Wrel, b_rel, W_out, b_out, out);
}
